// SwinTransformerBlock_43422119362772
// MI455X (gfx1250) — matrix operand order repaired and checked by reading, not compiled, not run on hardware
//
#include <hip/hip_runtime.h>

typedef __attribute__((ext_vector_type(16))) __bf16 bf16x16;
typedef __attribute__((ext_vector_type(8)))  float  f32x8;

#define MROWS 50176      // 16 * 56 * 56
#define CDIM  384
#define H3DIM 1152
#define HIDIM 1536

__device__ __forceinline__ unsigned short f2bf(float f) {
    union { float f; unsigned u; } v; v.f = f;
    unsigned r = v.u + 0x7FFFu + ((v.u >> 16) & 1u);
    return (unsigned short)(r >> 16);
}
__device__ __forceinline__ float bf2f(unsigned short h) {
    union { unsigned u; float f; } v; v.u = ((unsigned)h) << 16;
    return v.f;
}

// CDNA5 async DMA: global -> LDS, 16B per lane, tracked by ASYNCcnt.
// GVS mode: mem_addr = SADDR(sgpr64) + VADDR(u32, bytes); LDS dest = vdst vgpr.
__device__ __forceinline__ void async_ld_b128(unsigned lds_off, const void* sbase, unsigned voff) {
    asm volatile("global_load_async_to_lds_b128 %0, %1, %2"
                 :: "v"(lds_off), "v"(voff), "s"(sbase)
                 : "memory");
}
__device__ __forceinline__ void wait_async0() {
    asm volatile("s_wait_asynccnt 0x0" ::: "memory");
}
// low 32 bits of a generic LDS pointer == LDS byte offset (ISA aperture rule)
__device__ __forceinline__ unsigned lds_addr32(const void* p) {
    return (unsigned)(unsigned long long)p;
}

union FragU { uint4 u[2]; bf16x16 v; };

// A fragment (16x32 bf16): lane -> row = lane&15 ; K = kg*8..+7 and 16+kg*8..+7
__device__ __forceinline__ bf16x16 load_fragA(const unsigned short* rowPtr, int kg) {
    FragU r;
    const uint4* p = (const uint4*)rowPtr;
    r.u[0] = p[kg];
    r.u[1] = p[kg + 2];
    return r.v;
}
// B fragment (32x16 bf16) from K-contiguous (transposed) storage:
// lane -> col = lane&15 ; K = kg*8..+7 and 16+kg*8..+7
__device__ __forceinline__ bf16x16 load_fragB(const unsigned short* rowPtr, int kg) {
    FragU r;
    const uint4* p = (const uint4*)rowPtr;
    r.u[0] = p[kg];
    r.u[1] = p[kg + 2];
    return r.v;
}

// ---------------------------------------------------------------------------
// weight convert fp32 [K][N] -> bf16 transposed [N][K]
__global__ void wconv_kernel(const float* __restrict__ w, unsigned short* __restrict__ wt,
                             int K, int N) {
    int idx = blockIdx.x * 256 + threadIdx.x;
    if (idx >= K * N) return;
    int k = idx / N, n = idx - k * N;
    wt[(long)n * K + k] = f2bf(w[(long)k * N + n]);
}

// ---------------------------------------------------------------------------
// LayerNorm over C=384, one wave per row, output bf16
__global__ __launch_bounds__(256) void ln_kernel(const float* __restrict__ x,
                                                 const float* __restrict__ g,
                                                 const float* __restrict__ b,
                                                 unsigned short* __restrict__ out) {
    int row  = blockIdx.x * 8 + (threadIdx.x >> 5);
    int lane = threadIdx.x & 31;
    const float* xr = x + (long)row * CDIM;
    float v[12];
    float s = 0.f;
#pragma unroll
    for (int i = 0; i < 12; ++i) { v[i] = xr[lane + 32 * i]; s += v[i]; }
#pragma unroll
    for (int m = 16; m >= 1; m >>= 1) s += __shfl_xor(s, m, 32);
    float mu = s * (1.0f / 384.0f);
    float q = 0.f;
#pragma unroll
    for (int i = 0; i < 12; ++i) { float d = v[i] - mu; q += d * d; }
#pragma unroll
    for (int m = 16; m >= 1; m >>= 1) q += __shfl_xor(q, m, 32);
    float rstd = rsqrtf(q * (1.0f / 384.0f) + 1e-5f);
    unsigned short* orow = out + (long)row * CDIM;
#pragma unroll
    for (int i = 0; i < 12; ++i) {
        int c = lane + 32 * i;
        orow[c] = f2bf((v[i] - mu) * rstd * g[c] + b[c]);
    }
}

// ---------------------------------------------------------------------------
// bf16 WMMA GEMM: C[M,N] = A[M,K] * Bt[N,K]^T (+bias, epilogue)
// Operand tiles are DMA'd into LDS with global_load_async_to_lds_b128 and
// double-buffered: async engine fills buf[1-cur] while WMMAs consume buf[cur].
// EPI: 0 = bias -> bf16 ; 1 = bias + residual(f32) -> f32 ; 2 = bias + GELU -> bf16
template <int EPI>
__global__ __launch_bounds__(256)
void gemm_bf16(const unsigned short* __restrict__ A, const unsigned short* __restrict__ Bt,
               const float* __restrict__ bias, const float* __restrict__ res,
               unsigned short* __restrict__ outBf, float* __restrict__ outF,
               int M, int N, int K) {
    __shared__ __align__(16) unsigned short As[2][128][40];
    __shared__ __align__(16) unsigned short Bs[2][128][40];

    const int tid   = threadIdx.x;
    const int wave  = tid >> 5;
    const int lane  = tid & 31;
    const int waveM = wave & 3;          // 4 waves along M
    const int waveN = wave >> 2;         // 2 waves along N
    const int l15   = lane & 15;
    const int kg    = lane >> 4;
    const long mBase = (long)blockIdx.y * 128;
    const int  nBase = blockIdx.x * 128;
    const int  KT    = K >> 5;

    const unsigned short* Abase = A + mBase * K;            // uniform -> SGPR pair
    const unsigned short* Bbase = Bt + (long)nBase * K;

    auto issue = [&](int kt, int buf) {
#pragma unroll
        for (int cc = 0; cc < 2; ++cc) {
            int l2 = tid + cc * 256;
            int row = l2 >> 2, seg = l2 & 3;
            unsigned voff = (unsigned)((row * K + kt * 32 + seg * 8) * 2);
            async_ld_b128(lds_addr32(&As[buf][row][seg * 8]), Abase, voff);
            async_ld_b128(lds_addr32(&Bs[buf][row][seg * 8]), Bbase, voff);
        }
    };

    f32x8 acc[2][4] = {};
    issue(0, 0);
    wait_async0();
    __syncthreads();

    for (int kt = 0; kt < KT; ++kt) {
        const int cur = kt & 1;
        if (kt + 1 < KT) issue(kt + 1, 1 - cur);   // WAR safe: buf last read at kt-1
        if (kt + 2 < KT)
            __builtin_prefetch(Abase + (long)(tid >> 1) * K + (long)(kt + 2) * 32, 0, 0);

        bf16x16 af[2], bfm[4];
#pragma unroll
        for (int mt = 0; mt < 2; ++mt)
            af[mt] = load_fragA(&As[cur][waveM * 32 + mt * 16 + l15][0], kg);
#pragma unroll
        for (int nt = 0; nt < 4; ++nt)
            bfm[nt] = load_fragB(&Bs[cur][waveN * 64 + nt * 16 + l15][0], kg);
#pragma unroll
        for (int mt = 0; mt < 2; ++mt)
#pragma unroll
            for (int nt = 0; nt < 4; ++nt)
                acc[mt][nt] = __builtin_amdgcn_wmma_f32_16x16x32_bf16(
                    false, af[mt], false, bfm[nt], (short)0, acc[mt][nt], false, false);

        if (kt + 1 < KT) wait_async0();            // own async copies landed in LDS
        __syncthreads();                            // everyone's copies visible
    }

    // epilogue: C/D layout -> lane col = l15 ; rows = kg*8 + i
#pragma unroll
    for (int mt = 0; mt < 2; ++mt) {
        long gm0 = mBase + waveM * 32 + mt * 16 + kg * 8;
#pragma unroll
        for (int nt = 0; nt < 4; ++nt) {
            int gn = nBase + waveN * 64 + nt * 16 + l15;
            float bb = bias[gn];
#pragma unroll
            for (int i = 0; i < 8; ++i) {
                long gm = gm0 + i;
                float v = acc[mt][nt][i] + bb;
                if (EPI == 2) v = 0.5f * v * (1.0f + erff(v * 0.70710678118f));
                if (EPI == 1) {
                    v += res[gm * N + gn];
                    outF[gm * N + gn] = v;
                } else {
                    outBf[gm * N + gn] = f2bf(v);
                }
            }
        }
    }
}

// ---------------------------------------------------------------------------
// Attention: one block (128 thr, 4 waves) per (window, head). N=49 padded to 64.
__global__ __launch_bounds__(128)
void attn_kernel(const unsigned short* __restrict__ qkv, const float* __restrict__ rpb,
                 unsigned short* __restrict__ out) {
    __shared__ __align__(16) unsigned short qs[64][40];
    __shared__ __align__(16) unsigned short ks[64][40];
    __shared__ __align__(16) unsigned short vT[32][72];   // [d][kn]
    __shared__ __align__(16) unsigned short Ps[64][72];   // softmax probs bf16
    __shared__ int rowIdx[49];
    __shared__ int clsArr[49];

    const int w   = blockIdx.x;       // 0..1023
    const int h   = blockIdx.y;       // 0..11
    const int tid = threadIdx.x;

    // zero padded operand tiles
    for (int i = tid; i < 64 * 40 / 2; i += 128) {
        ((unsigned*)qs)[i] = 0u;
        ((unsigned*)ks)[i] = 0u;
    }
    for (int i = tid; i < 32 * 72 / 2; i += 128) ((unsigned*)vT)[i] = 0u;

    const int b  = w >> 6;
    const int wi = w & 63;
    const int wh = wi >> 3, ww = wi & 7;
    if (tid < 49) {
        int r = tid / 7, c = tid - r * 7;
        int hs = wh * 7 + r;             // rolled coords
        int wsC = ww * 7 + c;
        int ho = hs + 3;  if (ho >= 56) ho -= 56;   // undo roll(-3)
        int wo = wsC + 3; if (wo >= 56) wo -= 56;
        rowIdx[tid] = (b * 56 + ho) * 56 + wo;
        int ch = hs  < 49 ? 0 : (hs  < 53 ? 1 : 2);
        int cw = wsC < 49 ? 0 : (wsC < 53 ? 1 : 2);
        clsArr[tid] = ch * 3 + cw;
    }
    __syncthreads();

    // gather q (scaled, VGPR path), k (async DMA to LDS), v (transposed store)
    const float scale = 0.17677669529663687f;   // 1/sqrt(32)
    for (int idx = tid; idx < 49 * 4; idx += 128) {
        int n = idx >> 2, seg = idx & 3;
        long base = (long)rowIdx[n] * H3DIM + h * 32 + seg * 8;
        union { uint4 u; unsigned short e[8]; } t;
        t.u = *(const uint4*)(qkv + base);              // q
#pragma unroll
        for (int j = 0; j < 8; ++j) qs[n][seg * 8 + j] = f2bf(bf2f(t.e[j]) * scale);
        async_ld_b128(lds_addr32(&ks[n][seg * 8]), qkv, (unsigned)((base + CDIM) * 2)); // k
        t.u = *(const uint4*)(qkv + base + 2 * CDIM);   // v
#pragma unroll
        for (int j = 0; j < 8; ++j) vT[seg * 8 + j][n] = t.e[j];
    }
    wait_async0();
    __syncthreads();

    const int wave = tid >> 5, lane = tid & 31;
    const int l15 = lane & 15, kg = lane >> 4;

    // S = q * k^T  (wave owns 16 rows x 64 cols : 4 WMMAs)
    f32x8 s[4] = {};
    {
        bf16x16 aq = load_fragA(&qs[wave * 16 + l15][0], kg);
#pragma unroll
        for (int nt = 0; nt < 4; ++nt) {
            bf16x16 bk = load_fragB(&ks[nt * 16 + l15][0], kg);
            s[nt] = __builtin_amdgcn_wmma_f32_16x16x32_bf16(
                false, aq, false, bk, (short)0, s[nt], false, false);
        }
    }

    // bias + shift-mask, then row softmax (row lives in one half-wave)
    float vals[4][8];
    const int qnBase = wave * 16 + kg * 8;
#pragma unroll
    for (int nt = 0; nt < 4; ++nt) {
        int kn = nt * 16 + l15;
        bool kok = kn < 49;
        int kcls = kok ? clsArr[kn] : 0;
        int kr = kn / 7, kc = kn - kr * 7;
#pragma unroll
        for (int i = 0; i < 8; ++i) {
            int qn = qnBase + i;
            float v = s[nt][i];
            if (!kok) {
                v = -3.0e38f;
            } else if (qn < 49) {
                int qr = qn / 7, qc = qn - qr * 7;
                int rpi = (qr - kr + 6) * 13 + (qc - kc + 6);
                v += rpb[rpi * 12 + h];
                if (clsArr[qn] != kcls) v += -100.0f;
            }
            vals[nt][i] = v;
        }
    }
#pragma unroll
    for (int i = 0; i < 8; ++i) {
        float m = fmaxf(fmaxf(vals[0][i], vals[1][i]), fmaxf(vals[2][i], vals[3][i]));
#pragma unroll
        for (int msk = 1; msk < 16; msk <<= 1) m = fmaxf(m, __shfl_xor(m, msk, 32));
        float sum = 0.f;
#pragma unroll
        for (int nt = 0; nt < 4; ++nt) {
            float e = __expf(vals[nt][i] - m);
            vals[nt][i] = e;
            sum += e;
        }
#pragma unroll
        for (int msk = 1; msk < 16; msk <<= 1) sum += __shfl_xor(sum, msk, 32);
        float inv = 1.0f / sum;
#pragma unroll
        for (int nt = 0; nt < 4; ++nt)
            Ps[qnBase + i][nt * 16 + l15] = f2bf(vals[nt][i] * inv);
    }
    __syncthreads();

    // O = P * V : K=64 -> two k-steps, 2 n-tiles (hd=32)
    f32x8 o[2] = {};
#pragma unroll
    for (int ksi = 0; ksi < 2; ++ksi) {
        FragU ra2;
        const uint4* pr = (const uint4*)&Ps[wave * 16 + l15][0];  // pitch 72 el = 9 uint4
        ra2.u[0] = pr[ksi * 4 + kg];
        ra2.u[1] = pr[ksi * 4 + kg + 2];
#pragma unroll
        for (int nt = 0; nt < 2; ++nt) {
            FragU rb2;
            const uint4* pb = (const uint4*)&vT[nt * 16 + l15][0];
            rb2.u[0] = pb[ksi * 4 + kg];
            rb2.u[1] = pb[ksi * 4 + kg + 2];
            o[nt] = __builtin_amdgcn_wmma_f32_16x16x32_bf16(
                false, ra2.v, false, rb2.v, (short)0, o[nt], false, false);
        }
    }

    // scatter O back to original (rolled-back, window-reversed) row order
#pragma unroll
    for (int i = 0; i < 8; ++i) {
        int qn = qnBase + i;
        if (qn < 49) {
            long row = rowIdx[qn];
#pragma unroll
            for (int nt = 0; nt < 2; ++nt) {
                int d = nt * 16 + l15;
                out[row * CDIM + h * 32 + d] = f2bf(o[nt][i]);
            }
        }
    }
}

// ---------------------------------------------------------------------------
extern "C" void kernel_launch(void* const* d_in, const int* in_sizes, int n_in,
                              void* d_out, int out_size, void* d_ws, size_t ws_size,
                              hipStream_t stream) {
    const float* x      = (const float*)d_in[0];
    const float* n1g    = (const float*)d_in[1];
    const float* n1b    = (const float*)d_in[2];
    const float* qkv_w  = (const float*)d_in[3];
    const float* qkv_b  = (const float*)d_in[4];
    const float* rpb    = (const float*)d_in[5];
    const float* proj_w = (const float*)d_in[6];
    const float* proj_b = (const float*)d_in[7];
    const float* n2g    = (const float*)d_in[8];
    const float* n2b    = (const float*)d_in[9];
    const float* fc1_w  = (const float*)d_in[10];
    const float* fc1_b  = (const float*)d_in[11];
    const float* fc2_w  = (const float*)d_in[12];
    const float* fc2_b  = (const float*)d_in[13];
    (void)in_sizes; (void)n_in; (void)out_size; (void)ws_size;

    char* ws = (char*)d_ws;
    size_t off = 0;
    auto walloc = [&](size_t bytes) -> void* {
        void* p = ws + off;
        off += (bytes + 255) & ~(size_t)255;
        return p;
    };
    unsigned short* qkv_wT  = (unsigned short*)walloc((size_t)H3DIM * CDIM * 2);
    unsigned short* proj_wT = (unsigned short*)walloc((size_t)CDIM * CDIM * 2);
    unsigned short* fc1_wT  = (unsigned short*)walloc((size_t)HIDIM * CDIM * 2);
    unsigned short* fc2_wT  = (unsigned short*)walloc((size_t)CDIM * HIDIM * 2);
    unsigned short* xn      = (unsigned short*)walloc((size_t)MROWS * CDIM * 2);
    unsigned short* qkv     = (unsigned short*)walloc((size_t)MROWS * H3DIM * 2);
    unsigned short* attn    = (unsigned short*)walloc((size_t)MROWS * CDIM * 2);
    float*          x1      = (float*)walloc((size_t)MROWS * CDIM * 4);
    unsigned short* h2  = xn;    // alias: xn dead after QKV GEMM
    unsigned short* act = qkv;   // alias: qkv+attn regions (exactly MROWS*HIDIM*2) dead after proj
    float* out = (float*)d_out;

    // weights -> bf16, transposed [N][K]
    wconv_kernel<<<(CDIM * H3DIM + 255) / 256, 256, 0, stream>>>(qkv_w, qkv_wT, CDIM, H3DIM);
    wconv_kernel<<<(CDIM * CDIM + 255) / 256, 256, 0, stream>>>(proj_w, proj_wT, CDIM, CDIM);
    wconv_kernel<<<(CDIM * HIDIM + 255) / 256, 256, 0, stream>>>(fc1_w, fc1_wT, CDIM, HIDIM);
    wconv_kernel<<<(HIDIM * CDIM + 255) / 256, 256, 0, stream>>>(fc2_w, fc2_wT, HIDIM, CDIM);

    // LN1 -> bf16
    ln_kernel<<<MROWS / 8, 256, 0, stream>>>(x, n1g, n1b, xn);

    // QKV GEMM
    gemm_bf16<0><<<dim3(H3DIM / 128, MROWS / 128), 256, 0, stream>>>(
        xn, qkv_wT, qkv_b, nullptr, qkv, nullptr, MROWS, H3DIM, CDIM);

    // windowed attention (1024 windows x 12 heads)
    attn_kernel<<<dim3(1024, 12), 128, 0, stream>>>(qkv, rpb, attn);

    // proj GEMM + residual(x) -> x1 (f32)
    gemm_bf16<1><<<dim3(CDIM / 128, MROWS / 128), 256, 0, stream>>>(
        attn, proj_wT, proj_b, x, nullptr, x1, MROWS, CDIM, CDIM);

    // LN2 -> bf16
    ln_kernel<<<MROWS / 8, 256, 0, stream>>>(x1, n2g, n2b, h2);

    // fc1 GEMM + GELU -> bf16
    gemm_bf16<2><<<dim3(HIDIM / 128, MROWS / 128), 256, 0, stream>>>(
        h2, fc1_wT, fc1_b, nullptr, act, nullptr, MROWS, HIDIM, CDIM);

    // fc2 GEMM + residual(x1) -> out (f32)
    gemm_bf16<1><<<dim3(CDIM / 128, MROWS / 128), 256, 0, stream>>>(
        act, fc2_wT, fc2_b, x1, nullptr, out, MROWS, CDIM, HIDIM);
}
